// Encoder_20486994002518
// MI455X (gfx1250) — matrix operand order repaired and checked by reading, not compiled, not run on hardware
//
#include <hip/hip_runtime.h>
#include <stdint.h>

#define H 128

typedef __attribute__((ext_vector_type(16))) __bf16       v16bf;
typedef __attribute__((ext_vector_type(8)))  float        v8f;
typedef __attribute__((ext_vector_type(4)))  float        v4f;
typedef __attribute__((ext_vector_type(4)))  unsigned int v4u;

union AV { v16bf v; __bf16 b[16]; v4u q[2]; };

// ---------------------------------------------------------------------------
// Pack a row-major f32 weight matrix W[K x 128] into the per-lane WMMA B
// operand layout, bf16.  Chunk index c = (kt*8 + ct)*32 + lane, 16 bf16 each:
//   col = ct*16 + (lane&15);  K = kt*32 + (lane>>4)*8 + (j&7) + (j>>3)*16,  j = 0..15
// (dense 16-bit B 32x16: lanes 0-15 hold K 0..7,16..23, lanes 16-31 hold K 8..15,24..31)
// ---------------------------------------------------------------------------
__global__ __launch_bounds__(256) void pack_w_kernel(const float* __restrict__ W,
                                                     __bf16* __restrict__ out, int KT) {
    int t = blockIdx.x * 256 + threadIdx.x;
    int lane = t & 31, ct = (t >> 5) & 7, kt = t >> 8;
    if (kt >= KT) return;
    int col = ct * 16 + (lane & 15);
    int k0  = kt * 32 + (lane >> 4) * 8;
    __bf16* o = out + (size_t)t * 16;
#pragma unroll
    for (int j = 0; j < 16; j++) o[j] = (__bf16)(W[(size_t)(k0 + (j & 7) + (j >> 3) * 16) * H + col]);
}

// 8 WMMAs against the resident LDS weights, double-buffering the B tile so
// ds_load latency hides under the previous WMMA.
static __device__ __forceinline__ void wmma8_pf(const v4u* wsh, int kt, const v16bf a,
                                                int lane, v8f* acc) {
    const v4u* p = wsh + ((size_t)(kt * 8) * 32 + lane) * 2;
    AV b[2];
    b[0].q[0] = p[0]; b[0].q[1] = p[1];
#pragma unroll
    for (int ct = 0; ct < 8; ct++) {
        if (ct < 7) {
            b[(ct + 1) & 1].q[0] = p[(ct + 1) * 64];
            b[(ct + 1) & 1].q[1] = p[(ct + 1) * 64 + 1];
        }
        acc[ct] = __builtin_amdgcn_wmma_f32_16x16x32_bf16(
            false, a, false, b[ct & 1].v, (short)0, acc[ct], false, false);
    }
}

// ---------------------------------------------------------------------------
// Edge MLP: cat[g2m_e, grid[src], mesh[dst]] (K=384) -> silu -> (K=128) -> LN
// -> atomic scatter-add into agg[dst].
// Block = 256 thr = 8 waves; wave w owns rows [tile*128 + w*16, +16).
// All weights (W1 followed by W2, packed bf16) staged to LDS once per block;
// the tile loop itself is barrier-free (h staging is wave-private).
// ---------------------------------------------------------------------------
__global__ __launch_bounds__(256, 1)
void edge_mlp_kernel(const float* __restrict__ ef,
                     const float* __restrict__ grid,
                     const float* __restrict__ mesh,
                     const int* __restrict__ src,
                     const int* __restrict__ dst,
                     const v4u* __restrict__ Wp,  const float* __restrict__ b1,
                     const float* __restrict__ b2,
                     const float* __restrict__ g,  const float* __restrict__ bt,
                     float* __restrict__ agg, int NE, int ntiles) {
    extern __shared__ __align__(16) char smem[];            // 16 ktiles W + hsh
    v4u*    w1sh = (v4u*)smem;                              // 12 ktiles (96 KB)
    v4u*    w2sh = w1sh + 12 * 8 * 32 * 2;                  //  4 ktiles (32 KB)
    __bf16* hsh  = (__bf16*)(w2sh + 4 * 8 * 32 * 2);        // 8*16*128   (32 KB)

    const int tid = threadIdx.x;
    const int wave = tid >> 5, lane = tid & 31, lh = lane & 15, hi8 = (lane >> 4) * 8;
    const v8f zf = {0.f, 0.f, 0.f, 0.f, 0.f, 0.f, 0.f, 0.f};

    for (int i = tid; i < 16 * 8 * 32 * 2; i += 256) w1sh[i] = Wp[i];
    __syncthreads();

    __bf16* hrow = hsh + (size_t)(wave * 16 + lh) * H;       // GEMM2 A source
    __bf16* hst  = hsh + (size_t)wave * 16 * H;              // epilogue staging base

    for (int tile = blockIdx.x; tile < ntiles; tile += gridDim.x) {
        const int rowbase = tile * 128 + wave * 16;
        int myrow = rowbase + lh; if (myrow >= NE) myrow = NE - 1;
        const int si = src[myrow], di = dst[myrow];

        v8f acc[8];
#pragma unroll
        for (int i = 0; i < 8; i++) acc[i] = zf;

        for (int kt = 0; kt < 12; kt++) {                    // K = 384
            const float* ap;
            if      (kt < 4) ap = ef   + (size_t)myrow * H +  kt      * 32;
            else if (kt < 8) ap = grid + (size_t)si    * H + (kt - 4) * 32;
            else             ap = mesh + (size_t)di    * H + (kt - 8) * 32;
            v4f f0 = *(const v4f*)(ap + hi8);
            v4f f1 = *(const v4f*)(ap + hi8 + 4);
            v4f f2 = *(const v4f*)(ap + hi8 + 16);
            v4f f3 = *(const v4f*)(ap + hi8 + 20);
            AV a;
#pragma unroll
            for (int j = 0; j < 4; j++) {
                a.b[j]      = (__bf16)f0[j];  a.b[4 + j]  = (__bf16)f1[j];
                a.b[8 + j]  = (__bf16)f2[j];  a.b[12 + j] = (__bf16)f3[j];
            }
            wmma8_pf(w1sh, kt, a.v, lane, acc);
        }

        // bias + SiLU, stage h (bf16, row-major) for GEMM2's A operand
#pragma unroll
        for (int ct = 0; ct < 8; ct++) {
            float bb = b1[ct * 16 + lh];
#pragma unroll
            for (int v = 0; v < 8; v++) {
                float x = acc[ct][v] + bb;
                float h = x / (1.0f + __expf(-x));
                hst[(size_t)(v + hi8) * H + ct * 16 + lh] = (__bf16)h;
            }
        }

        v8f acc2[8];
#pragma unroll
        for (int i = 0; i < 8; i++) acc2[i] = zf;
        for (int kt = 0; kt < 4; kt++) {                     // K = 128
            AV a;
            a.q[0] = *(const v4u*)&hrow[kt * 32 + hi8];
            a.q[1] = *(const v4u*)&hrow[kt * 32 + hi8 + 16];
            wmma8_pf(w2sh, kt, a.v, lane, acc2);
        }

        // hoisted epilogue constants (per lane: one column of each col-tile)
        float b2v[8], gv[8], btv[8];
#pragma unroll
        for (int ct = 0; ct < 8; ct++) {
            b2v[ct] = b2[ct * 16 + lh];
            gv[ct]  = g[ct * 16 + lh];
            btv[ct] = bt[ct * 16 + lh];
        }

        // bias + LayerNorm (row = VGPR index, 128 cols over half-wave lanes x
        // 8 col-tiles), then scatter-add to agg[dst].
#pragma unroll
        for (int v = 0; v < 8; v++) {
            float x[8], s1 = 0.f, s2 = 0.f;
#pragma unroll
            for (int ct = 0; ct < 8; ct++) {
                float t = acc2[ct][v] + b2v[ct];
                x[ct] = t; s1 += t; s2 += t * t;
            }
#pragma unroll
            for (int m = 1; m < 16; m <<= 1) {
                s1 += __shfl_xor(s1, m, 32);
                s2 += __shfl_xor(s2, m, 32);
            }
            float mean = s1 * (1.0f / 128.0f);
            float var  = s2 * (1.0f / 128.0f) - mean * mean;
            float inv  = rsqrtf(var + 1e-5f);
            int r = rowbase + v + hi8;
            if (r < NE) {
                float* ag = agg + (size_t)dst[r] * H;
#pragma unroll
                for (int ct = 0; ct < 8; ct++) {
                    float y = (x[ct] - mean) * inv * gv[ct] + btv[ct];
                    unsafeAtomicAdd(ag + ct * 16 + lh, y);
                }
            }
        }
    }
}

// ---------------------------------------------------------------------------
// Node MLP (src: KT1=4, in0=grid; dst: KT1=8, in0=agg, in1=mesh), LN,
// residual add, direct store.
// ---------------------------------------------------------------------------
template <int KT1>
__global__ __launch_bounds__(256, 1)
void node_mlp_kernel(const float* __restrict__ in0,
                     const float* __restrict__ in1,
                     const float* __restrict__ resid,
                     const v4u* __restrict__ Wp,  const float* __restrict__ b1,
                     const float* __restrict__ b2,
                     const float* __restrict__ g,  const float* __restrict__ bt,
                     float* __restrict__ out, int Nrows, int ntiles) {
    extern __shared__ __align__(16) char smem[];
    v4u*    w1sh = (v4u*)smem;                               // KT1 ktiles
    v4u*    w2sh = w1sh + KT1 * 8 * 32 * 2;                  // 4 ktiles
    __bf16* hsh  = (__bf16*)(w2sh + 4 * 8 * 32 * 2);         // 8*16*128

    const int tid = threadIdx.x;
    const int wave = tid >> 5, lane = tid & 31, lh = lane & 15, hi8 = (lane >> 4) * 8;
    const v8f zf = {0.f, 0.f, 0.f, 0.f, 0.f, 0.f, 0.f, 0.f};

    for (int i = tid; i < (KT1 + 4) * 8 * 32 * 2; i += 256) w1sh[i] = Wp[i];
    __syncthreads();

    __bf16* hrow = hsh + (size_t)(wave * 16 + lh) * H;
    __bf16* hst  = hsh + (size_t)wave * 16 * H;

    for (int tile = blockIdx.x; tile < ntiles; tile += gridDim.x) {
        const int rowbase = tile * 128 + wave * 16;
        int myrow = rowbase + lh; if (myrow >= Nrows) myrow = Nrows - 1;

        v8f acc[8];
#pragma unroll
        for (int i = 0; i < 8; i++) acc[i] = zf;

        for (int kt = 0; kt < KT1; kt++) {
            const float* ap;
            if (KT1 == 4 || kt < 4) ap = in0 + (size_t)myrow * H + kt * 32;
            else                    ap = in1 + (size_t)myrow * H + (kt - 4) * 32;
            v4f f0 = *(const v4f*)(ap + hi8);
            v4f f1 = *(const v4f*)(ap + hi8 + 4);
            v4f f2 = *(const v4f*)(ap + hi8 + 16);
            v4f f3 = *(const v4f*)(ap + hi8 + 20);
            AV a;
#pragma unroll
            for (int j = 0; j < 4; j++) {
                a.b[j]      = (__bf16)f0[j];  a.b[4 + j]  = (__bf16)f1[j];
                a.b[8 + j]  = (__bf16)f2[j];  a.b[12 + j] = (__bf16)f3[j];
            }
            wmma8_pf(w1sh, kt, a.v, lane, acc);
        }

#pragma unroll
        for (int ct = 0; ct < 8; ct++) {
            float bb = b1[ct * 16 + lh];
#pragma unroll
            for (int v = 0; v < 8; v++) {
                float x = acc[ct][v] + bb;
                float h = x / (1.0f + __expf(-x));
                hst[(size_t)(v + hi8) * H + ct * 16 + lh] = (__bf16)h;
            }
        }

        v8f acc2[8];
#pragma unroll
        for (int i = 0; i < 8; i++) acc2[i] = zf;
        for (int kt = 0; kt < 4; kt++) {
            AV a;
            a.q[0] = *(const v4u*)&hrow[kt * 32 + hi8];
            a.q[1] = *(const v4u*)&hrow[kt * 32 + hi8 + 16];
            wmma8_pf(w2sh, kt, a.v, lane, acc2);
        }

        float b2v[8], gv[8], btv[8];
#pragma unroll
        for (int ct = 0; ct < 8; ct++) {
            b2v[ct] = b2[ct * 16 + lh];
            gv[ct]  = g[ct * 16 + lh];
            btv[ct] = bt[ct * 16 + lh];
        }

#pragma unroll
        for (int v = 0; v < 8; v++) {
            float x[8], s1 = 0.f, s2 = 0.f;
#pragma unroll
            for (int ct = 0; ct < 8; ct++) {
                float t = acc2[ct][v] + b2v[ct];
                x[ct] = t; s1 += t; s2 += t * t;
            }
#pragma unroll
            for (int m = 1; m < 16; m <<= 1) {
                s1 += __shfl_xor(s1, m, 32);
                s2 += __shfl_xor(s2, m, 32);
            }
            float mean = s1 * (1.0f / 128.0f);
            float var  = s2 * (1.0f / 128.0f) - mean * mean;
            float inv  = rsqrtf(var + 1e-5f);
            int r = rowbase + v + hi8;
            if (r < Nrows) {
                const float* rp = resid + (size_t)r * H;
                float*       op = out   + (size_t)r * H;
#pragma unroll
                for (int ct = 0; ct < 8; ct++) {
                    int col = ct * 16 + lh;
                    float y = (x[ct] - mean) * inv * gv[ct] + btv[ct];
                    op[col] = rp[col] + y;
                }
            }
        }
    }
}

// ---------------------------------------------------------------------------
extern "C" void kernel_launch(void* const* d_in, const int* in_sizes, int n_in,
                              void* d_out, int out_size, void* d_ws, size_t ws_size,
                              hipStream_t stream) {
    const float* ef   = (const float*)d_in[0];
    const float* grid = (const float*)d_in[1];
    const float* mesh = (const float*)d_in[2];
    const int*   src  = (const int*)d_in[3];
    const int*   dst  = (const int*)d_in[4];
    const float *eW1 = (const float*)d_in[6],  *eb1 = (const float*)d_in[7];
    const float *eW2 = (const float*)d_in[8],  *eb2 = (const float*)d_in[9];
    const float *eg  = (const float*)d_in[10], *ebt = (const float*)d_in[11];
    const float *sW1 = (const float*)d_in[12], *sb1 = (const float*)d_in[13];
    const float *sW2 = (const float*)d_in[14], *sb2 = (const float*)d_in[15];
    const float *sg  = (const float*)d_in[16], *sbt = (const float*)d_in[17];
    const float *dW1 = (const float*)d_in[18], *db1 = (const float*)d_in[19];
    const float *dW2 = (const float*)d_in[20], *db2 = (const float*)d_in[21];
    const float *dg  = (const float*)d_in[22], *dbt = (const float*)d_in[23];

    const int NE = in_sizes[3];
    const int NG = in_sizes[1] / H;
    const int NM = in_sizes[2] / H;
    float* out = (float*)d_out;                 // [NG*H] grid_new, then [NM*H] mesh_new

    // workspace carve: agg (f32, NM x H) then six packed bf16 weight blobs.
    // NOTE: each kernel's W1 blob is immediately followed by its W2 blob so a
    // single contiguous LDS copy covers both.
    char*  w   = (char*)d_ws;
    float* agg = (float*)w;
    size_t off = ((size_t)NM * H * 4 + 255) & ~(size_t)255;
    auto cbytes = [](int KT) { return (size_t)KT * 8 * 32 * 32; };
    v4u* eW1p = (v4u*)(w + off); off += cbytes(12);
    v4u* eW2p = (v4u*)(w + off); off += cbytes(4);
    v4u* sW1p = (v4u*)(w + off); off += cbytes(4);
    v4u* sW2p = (v4u*)(w + off); off += cbytes(4);
    v4u* dW1p = (v4u*)(w + off); off += cbytes(8);
    v4u* dW2p = (v4u*)(w + off); off += cbytes(4);

    pack_w_kernel<<<12, 256, 0, stream>>>(eW1, (__bf16*)eW1p, 12);
    pack_w_kernel<<< 4, 256, 0, stream>>>(eW2, (__bf16*)eW2p,  4);
    pack_w_kernel<<< 4, 256, 0, stream>>>(sW1, (__bf16*)sW1p,  4);
    pack_w_kernel<<< 4, 256, 0, stream>>>(sW2, (__bf16*)sW2p,  4);
    pack_w_kernel<<< 8, 256, 0, stream>>>(dW1, (__bf16*)dW1p,  8);
    pack_w_kernel<<< 4, 256, 0, stream>>>(dW2, (__bf16*)dW2p,  4);

    hipMemsetAsync(agg, 0, (size_t)NM * H * 4, stream);

    const int ntE = (NE + 127) / 128;
    const int ntG = (NG + 127) / 128;
    const int ntM = (NM + 127) / 128;

    const size_t ldsE = (size_t)(12 + 4) * 8 * 32 * 32 + 8 * 16 * H * 2;  // 160 KB
    const size_t ldsD = (size_t)( 8 + 4) * 8 * 32 * 32 + 8 * 16 * H * 2;  // 128 KB
    const size_t ldsS = (size_t)( 4 + 4) * 8 * 32 * 32 + 8 * 16 * H * 2;  //  96 KB

    edge_mlp_kernel<<<2048, 256, ldsE, stream>>>(ef, grid, mesh, src, dst,
                                                 eW1p, eb1, eb2, eg, ebt,
                                                 agg, NE, ntE);
    node_mlp_kernel<8><<<ntM, 256, ldsD, stream>>>(agg, mesh, mesh,
                                                   dW1p, db1, db2, dg, dbt,
                                                   out + (size_t)NG * H, NM, ntM);
    node_mlp_kernel<4><<<2048, 256, ldsS, stream>>>(grid, nullptr, grid,
                                                    sW1p, sb1, sb2, sg, sbt,
                                                    out, NG, ntG);
}
